// CRF_9904194584633
// MI455X (gfx1250) — matrix operand order repaired and checked by reading, not compiled, not run on hardware
//
#include <hip/hip_runtime.h>
#include <hip/hip_fp16.h>

typedef __attribute__((ext_vector_type(16))) _Float16 v16h;
typedef __attribute__((ext_vector_type(8)))  _Float16 v8h;
typedef __attribute__((ext_vector_type(8)))  float    v8f;
typedef __attribute__((ext_vector_type(4)))  float    v4f;
typedef __attribute__((ext_vector_type(4)))  unsigned int v4u;
typedef __attribute__((ext_vector_type(8)))  int      v8i_t;
typedef __attribute__((ext_vector_type(4)))  int      v4i_t;

constexpr int   B_      = 128;
constexpr int   S_      = 256;
constexpr int   L_      = 512;
constexpr int   BT      = 16;     // batch rows per workgroup
constexpr float NEG_INF = -10000.0f;

// dynamic LDS carve (bytes)
constexpr unsigned OFF_ALPHA = 0;                         // float [16][512]  32 KB
constexpr unsigned OFF_LOG   = OFF_ALPHA + BT * L_ * 4;   // float [2][16][512] 64 KB
constexpr unsigned OFF_EA    = OFF_LOG + 2 * BT * L_ * 4; // f16 [16][512]   16 KB
constexpr unsigned OFF_MS    = OFF_EA + BT * L_ * 2;      // float [16]
constexpr unsigned SMEM_BYTES = OFF_MS + BT * 4;

#if __has_builtin(__builtin_amdgcn_tensor_load_to_lds) && \
    __has_builtin(__builtin_amdgcn_s_wait_tensorcnt)
#define USE_TDM 1
#else
#define USE_TDM 0
#endif

// ---------------------------------------------------------------------------
__global__ void crf_prep_expT(const float* __restrict__ T,
                              _Float16* __restrict__ W) {
    int idx = blockIdx.x * blockDim.x + threadIdx.x;
    if (idx < L_ * L_) W[idx] = (_Float16)__expf(T[idx]);
}

#if USE_TDM
// TDM load of a 2-D tile: 16 rows x 512 f32, row stride S*L floats, into LDS.
// D# layout per CDNA5 ISA §8.3/§8.4; 6-arg builtin (clang-23/therock-10.0).
__device__ __forceinline__ void tdm_load_tile(const float* gptr,
                                              unsigned lds_off) {
    unsigned long long ga = (unsigned long long)(uintptr_t)gptr;
    v4u g0;
    g0.x = 1u;                                        // count=1, user mode
    g0.y = lds_off;                                   // lds_addr
    g0.z = (unsigned)(ga & 0xffffffffu);              // global_addr[31:0]
    g0.w = (unsigned)((ga >> 32) & 0x01ffffffu) | (2u << 30); // ga[56:32]|type=2
    v8i_t g1;
    g1[0] = (int)(2u << 16);        // data_size=4B; mask=0 (not in cluster)
    g1[1] = (int)(512u << 16);      // tensor_dim0 = 512 (lo16 @ bit48)
    g1[2] = (int)(16u << 16);       // tensor_dim0 hi=0 | tensor_dim1 = 16
    g1[3] = (int)(512u << 16);      // tensor_dim1 hi=0 | tile_dim0 = 512
    g1[4] = 16;                     // tile_dim1 = 16, tile_dim2 = 0
    g1[5] = S_ * L_;                // tensor_dim0_stride = 131072 (lo32)
    g1[6] = 0;                      // stride hi | tensor_dim1_stride lo
    g1[7] = 0;
    v4i_t z4 = {0, 0, 0, 0};        // groups 2/3 unused (2-D tensor)
    v8i_t z8 = {0, 0, 0, 0, 0, 0, 0, 0};
    __builtin_amdgcn_tensor_load_to_lds(g0, g1, z4, z4, z8, 0);
}
#endif

// ---------------------------------------------------------------------------
// 512 threads = 16 wave32 waves per workgroup; 8 workgroups (16 batch rows ea).
// Wave w: (a) owns reductions for batch row w; (b) owns label columns
// [32w, 32w+32) as two 16x16 WMMA tiles with the B operand (exp(T) slice)
// held register-resident across all 256 scan steps.
// ---------------------------------------------------------------------------
__global__ __launch_bounds__(512)
void crf_forward(const float* __restrict__ logits,
                 const int*   __restrict__ lens,
                 const float* __restrict__ T,
                 const _Float16* __restrict__ W,
                 float* __restrict__ out) {
    extern __shared__ char smem[];
    float*    alphaS = (float*)(smem + OFF_ALPHA);   // [16][512]
    float*    logS   = (float*)(smem + OFF_LOG);     // [2][16][512]
    _Float16* eaS    = (_Float16*)(smem + OFF_EA);   // [16][512]
    float*    mS     = (float*)(smem + OFF_MS);      // [16]

    const int tid  = threadIdx.x;
    const int lane = tid & 31;
    const int wave = tid >> 5;          // 0..15: reduction row AND column pair
    const int half = lane >> 4;         // ISA lane-group (VGPR striping)
    const int mm   = lane & 15;         // A row / B,D column within a tile
    const int c0   = wave * 32;         // first owned column tile
    const int b0   = blockIdx.x * BT;

    // ---- loop-invariant register state ----------------------------------
    int lenR[8];
#pragma unroll
    for (int r = 0; r < 8; ++r) lenR[r] = lens[b0 + r + 8 * half];

    // B operand register-resident: B[k,n] = expT[n,k]; lane holds column
    // n = ct + mm, elements K = k0 + 16*half + e  (contiguous 32B loads).
    v16h Breg0[16], Breg1[16];
#pragma unroll
    for (int kt = 0; kt < 16; ++kt) {
        Breg0[kt] = *(const v16h*)&W[(size_t)(c0 + mm) * L_ + kt * 32 + 16 * half];
        Breg1[kt] = *(const v16h*)&W[(size_t)(c0 + 16 + mm) * L_ + kt * 32 + 16 * half];
    }

    // alpha tile cache (D layout: M = r + 8*half, N = mm)
    float aReg[2][8];
#pragma unroll
    for (int tc = 0; tc < 2; ++tc) {
        const float v = (c0 + 16 * tc + mm == L_ - 2) ? 0.0f : NEG_INF;
#pragma unroll
        for (int r = 0; r < 8; ++r) aReg[tc][r] = v;
    }

    // ---- init alpha in LDS (wave w = row w, contiguous 16 cols/lane) ----
#pragma unroll
    for (int e = 0; e < 16; ++e) {
        const int col = lane * 16 + e;
        alphaS[wave * L_ + col] = (col == L_ - 2) ? 0.0f : NEG_INF;
    }
#if USE_TDM
    if (wave == 0) {
        unsigned logS_off = (unsigned)(uintptr_t)logS;   // low 32 = LDS addr
        tdm_load_tile(logits + (size_t)b0 * S_ * L_, logS_off);
    }
#endif
    __syncthreads();

    for (int t = 0; t < S_; ++t) {
        // ---- phase 1+2: row max + ea = exp(alpha-m), wave w == row w ----
        // contiguous 16 columns per lane -> 4x ds_load_b128, 2x ds_store_b128
        v4f x[4];
#pragma unroll
        for (int q = 0; q < 4; ++q)
            x[q] = *(const v4f*)&alphaS[wave * L_ + lane * 16 + 4 * q];
        float m = NEG_INF;
#pragma unroll
        for (int q = 0; q < 4; ++q)
#pragma unroll
            for (int e = 0; e < 4; ++e) m = fmaxf(m, x[q][e]);
#pragma unroll
        for (int off = 16; off > 0; off >>= 1)
            m = fmaxf(m, __shfl_xor(m, off, 32));
        v8h h0, h1;
#pragma unroll
        for (int e = 0; e < 8; ++e) {
            h0[e] = (_Float16)__expf(x[e >> 2][e & 3] - m);
            h1[e] = (_Float16)__expf(x[2 + (e >> 2)][e & 3] - m);
        }
        *(v8h*)&eaS[wave * L_ + lane * 16]     = h0;
        *(v8h*)&eaS[wave * L_ + lane * 16 + 8] = h1;
        if (lane == 0) mS[wave] = m;

#if USE_TDM
        if (wave == 0) {
            if (t + 1 < S_) {
                unsigned logS_off = (unsigned)(uintptr_t)logS;
                tdm_load_tile(logits + ((size_t)b0 * S_ + (t + 1)) * L_,
                              logS_off + ((unsigned)((t + 1) & 1)) * (BT * L_ * 4));
                __builtin_amdgcn_s_wait_tensorcnt(1);  // in-order: tile(t) done
            } else {
                __builtin_amdgcn_s_wait_tensorcnt(0);
            }
        }
#endif
        __syncthreads();   // publishes eaS, mS (and TDM tile t)

        // ---- phase 3: two 16x16 tiles, B register-resident --------------
        // minimal A live range: load -> shuffle -> 2x wmma per kt
        v8f cAcc[2];
        cAcc[0] = (v8f){0.f, 0.f, 0.f, 0.f, 0.f, 0.f, 0.f, 0.f};
        cAcc[1] = (v8f){0.f, 0.f, 0.f, 0.f, 0.f, 0.f, 0.f, 0.f};
#pragma unroll
        for (int kt = 0; kt < 16; ++kt) {
            const int k0 = kt * 32;
            // A 16x32 f16: K(e) = 16*half + e
            const v8h a0 = *(const v8h*)&eaS[mm * L_ + k0 + 16 * half];
            const v8h a1 = *(const v8h*)&eaS[mm * L_ + k0 + 16 * half + 8];
            const v16h a = __builtin_shufflevector(
                a0, a1, 0, 1, 2, 3, 4, 5, 6, 7, 8, 9, 10, 11, 12, 13, 14, 15);
            cAcc[0] = __builtin_amdgcn_wmma_f32_16x16x32_f16(
                false, a, false, Breg0[kt], (short)0, cAcc[0], false, false);
            cAcc[1] = __builtin_amdgcn_wmma_f32_16x16x32_f16(
                false, a, false, Breg1[kt], (short)0, cAcc[1], false, false);
        }

        // ---- phase 4: branchless masked alpha update --------------------
#pragma unroll
        for (int tc = 0; tc < 2; ++tc) {
            const int i = c0 + 16 * tc + mm;
#pragma unroll
            for (int r = 0; r < 8; ++r) {
                const int M = r + 8 * half;
#if USE_TDM
                const float lg = logS[(t & 1) * (BT * L_) + M * L_ + i];
#else
                const float lg = logits[((size_t)(b0 + M) * S_ + t) * L_ + i];
#endif
                const float vnew = __logf(cAcc[tc][r]) + mS[M] + lg;
                const float val  = (t < lenR[r]) ? vnew : aReg[tc][r];
                aReg[tc][r] = val;
                alphaS[M * L_ + i] = val;
            }
        }
        __syncthreads();   // publishes alphaS for next step's phase 1
    }

    // ---- epilogue: alpha += T[stop,:]; out = logsumexp per row ----------
    {
        float vloc[16];
        float m = -3.4e38f;
#pragma unroll
        for (int e = 0; e < 16; ++e) {
            const int col = lane * 16 + e;
            vloc[e] = alphaS[wave * L_ + col] + T[(size_t)(L_ - 1) * L_ + col];
            m = fmaxf(m, vloc[e]);
        }
#pragma unroll
        for (int off = 16; off > 0; off >>= 1)
            m = fmaxf(m, __shfl_xor(m, off, 32));
        float s = 0.f;
#pragma unroll
        for (int e = 0; e < 16; ++e) s += __expf(vloc[e] - m);
#pragma unroll
        for (int off = 16; off > 0; off >>= 1)
            s += __shfl_xor(s, off, 32);
        if (lane == 0) out[b0 + wave] = __logf(s) + m;
    }
}

// ---------------------------------------------------------------------------
extern "C" void kernel_launch(void* const* d_in, const int* in_sizes, int n_in,
                              void* d_out, int out_size, void* d_ws,
                              size_t ws_size, hipStream_t stream) {
    (void)in_sizes; (void)n_in; (void)out_size; (void)ws_size;
    const float* logits = (const float*)d_in[0];
    const int*   lens   = (const int*)d_in[1];
    const float* trans  = (const float*)d_in[2];
    float*       out    = (float*)d_out;
    _Float16*    W      = (_Float16*)d_ws;   // 512 KB scratch: exp(T) in f16

    crf_prep_expT<<<(L_ * L_ + 255) / 256, 256, 0, stream>>>(trans, W);
    crf_forward<<<B_ / BT, 512, SMEM_BYTES, stream>>>(logits, lens, trans, W, out);
}
